// MultiModalAttentionV3_85641647882455
// MI455X (gfx1250) — matrix operand order repaired and checked by reading, not compiled, not run on hardware
//
#include <hip/hip_runtime.h>
#include <hip/hip_bf16.h>

#define TDM_PATH 1   // 1: Tensor Data Mover staging + double buffering; 0: per-thread LDS staging

// ---------------- types ----------------
typedef float        v8f   __attribute__((ext_vector_type(8)));
typedef __bf16       v16bf __attribute__((ext_vector_type(16)));
typedef unsigned int u32x4 __attribute__((ext_vector_type(4)));
typedef unsigned int u32x8 __attribute__((ext_vector_type(8)));

// ---------------- problem constants ----------------
constexpr int kB = 2, kT = 2048, kM = 512, kC = 512, kH = 8, kD = kC / kH; // D = 64
constexpr float kSCALE = 0.125f; // 1/sqrt(64)

// =====================================================================
// TDM: issue a 2D tile load (rows x 32 bf16 elements, row stride ldElems)
// into LDS at ldsAddr, padding each 64B row with 16B so the LDS row stride
// is 80B (= 40 bf16), matching the WMMA fragment addressing below.
// Descriptor packing per CDNA5 ISA (D# groups 0/1).
// =====================================================================
__device__ __forceinline__ void tdm_load_tile(unsigned ldsAddr, const void* gptr,
                                              int rows, long ldElems) {
  union { unsigned u[4]; u32x4 v; } g0;
  const unsigned long long ga = (unsigned long long)(size_t)gptr;
  g0.u[0] = 1u;                                                   // count = 1 valid descriptor
  g0.u[1] = ldsAddr;                                              // lds_addr (bytes)
  g0.u[2] = (unsigned)ga;                                         // global_addr[31:0]
  g0.u[3] = (unsigned)((ga >> 32) & 0x01FFFFFFu) | (2u << 30);    // global_addr[56:32] | type=2

  union { unsigned long long q[4]; u32x8 v; } g1;
  const unsigned long long stride = (unsigned long long)ldElems;  // tensor_dim0_stride (elems)
  const unsigned long long tdim0  = (unsigned long long)(unsigned long long)ldElems;
  g1.q[0] = (1ull << 16)                       // data_size = 1 -> 2 bytes (bf16)
          | (1ull << 20)                       // pad_enable
          | (3ull << 22)                       // pad_interval code 3 -> every 16 DWORDs (64B)
          | (3ull << 25)                       // pad_amount   code 3 -> 4 DWORDs (16B)
          | ((tdim0 & 0xFFFFull) << 48);       // tensor_dim0[15:0]
  g1.q[1] = ((tdim0 >> 16) & 0xFFFFull)        // tensor_dim0[31:16]
          | (0x100000ull << 16)                // tensor_dim1 = 1<<20 (covers all rows)
          | (32ull << 48);                     // tile_dim0 = 32 elements (one K-step)
  g1.q[2] = (unsigned long long)(rows & 0xFFFF)        // tile_dim1 = rows (tile_dim2 = 0)
          | ((stride & 0xFFFFFFFFull) << 32);          // tensor_dim0_stride[31:0]
  g1.q[3] = (stride >> 32) & 0xFFFFull;                // tensor_dim0_stride[47:32]

  asm volatile("tensor_load_to_lds %0, %1" :: "s"(g0.v), "s"(g1.v) : "memory");
}

// =====================================================================
// Generic WMMA GEMM:  D[row, col] = alpha * sum_k A[row,k] * BT[col,k] + bias[col]
// A, BT row-major bf16.  Block tile 128x64, K-step 32, 8 wave32s per WG,
// each wave computes 32x32 via 4x v_wmma_f32_16x16x32_bf16.
// Staging: TDM tensor_load_to_lds with double-buffered LDS (wave 0 issues,
// waits on TENSORcnt; workgroup barrier publishes to all waves).
// =====================================================================
__global__ __launch_bounds__(256) void wmma_gemm_bt(
    const __bf16* __restrict__ A, const __bf16* __restrict__ BT,
    const float* __restrict__ bias, float alpha,
    float* __restrict__ Df32, __bf16* __restrict__ Dbf16,
    int K, long ldA, long ldBT, long ldD,
    long bsA, long bsBT, long dOuter, long dInner, int dInnerCnt)
{
  const int  tid  = threadIdx.x;
  const int  z    = blockIdx.z;
  const long rowBase = (long)blockIdx.x * 128;
  const long colBase = (long)blockIdx.y * 64;

  A  += (long)z * bsA;
  BT += (long)z * bsBT;
  const long dOff = (long)(z / dInnerCnt) * dOuter + (long)(z % dInnerCnt) * dInner;

  const int wid  = tid >> 5;
  const int lane = tid & 31;
  const int wm   = wid >> 1;   // 0..3 -> M offset wm*32
  const int wn   = wid & 1;    // 0..1 -> N offset wn*32
  const int hi   = lane >> 4;
  const int lo   = lane & 15;

  v8f acc[2][2] = {};

#if TDM_PATH
  __shared__ __bf16 lsA[2][128 * 40];   // 80B LDS row stride via TDM padding
  __shared__ __bf16 lsB[2][64 * 40];

  const __bf16* Abase = A  + rowBase * ldA;
  const __bf16* Bbase = BT + colBase * ldBT;
  const unsigned aAddr[2] = { (unsigned)(size_t)&lsA[0][0], (unsigned)(size_t)&lsA[1][0] };
  const unsigned bAddr[2] = { (unsigned)(size_t)&lsB[0][0], (unsigned)(size_t)&lsB[1][0] };

  if (wid == 0) {                      // stage 0 prologue
    tdm_load_tile(aAddr[0], Abase, 128, ldA);
    tdm_load_tile(bAddr[0], Bbase, 64,  ldBT);
  }

  int buf = 0;
  for (int k0 = 0; k0 < K; k0 += 32) {
    if (wid == 0) {
      if (k0 + 32 < K) {               // prefetch next stage into other buffer
        tdm_load_tile(aAddr[buf ^ 1], Abase + (k0 + 32), 128, ldA);
        tdm_load_tile(bAddr[buf ^ 1], Bbase + (k0 + 32), 64,  ldBT);
        __builtin_amdgcn_s_wait_tensorcnt(2);  // current stage done; next may fly
      } else {
        __builtin_amdgcn_s_wait_tensorcnt(0);
      }
    }
    __syncthreads();                   // publish TDM-written tiles to all waves

    const __bf16* cA = &lsA[buf][0];
    const __bf16* cB = &lsB[buf][0];
#else
  __shared__ __bf16 lsA1[128 * 40];
  __shared__ __bf16 lsB1[64 * 40];
  const int r0 = tid >> 2;            // 0..63
  const int c0 = (tid & 3) * 8;       // 0,8,16,24
  for (int k0 = 0; k0 < K; k0 += 32) {
    *(uint4*)&lsA1[(r0     ) * 40 + c0] = *(const uint4*)&A[(rowBase + r0     ) * ldA + k0 + c0];
    *(uint4*)&lsA1[(r0 + 64) * 40 + c0] = *(const uint4*)&A[(rowBase + r0 + 64) * ldA + k0 + c0];
    *(uint4*)&lsB1[(r0     ) * 40 + c0] = *(const uint4*)&BT[(colBase + r0    ) * ldBT + k0 + c0];
    __syncthreads();
    const __bf16* cA = lsA1;
    const __bf16* cB = lsB1;
#endif

    // ---- build fragments per ISA 16-bit A(16x32)/B(32x16) layouts ----
    union Frag { v16bf v; uint4 q[2]; };
    Frag a[2], b[2];
#pragma unroll
    for (int i = 0; i < 2; ++i) {
      const int r  = wm * 32 + i * 16 + lo;   // A row
      const int kb = hi * 8;                  // K base (0 or 8)
      a[i].q[0] = *(const uint4*)&cA[r * 40 + kb];       // K = kb .. kb+7
      a[i].q[1] = *(const uint4*)&cA[r * 40 + kb + 16];  // K = kb+16 .. kb+23
    }
#pragma unroll
    for (int j = 0; j < 2; ++j) {
      const int r  = wn * 32 + j * 16 + lo;   // BT row == output col
      const int kb = hi * 8;                  // K base (0 or 8)
      b[j].q[0] = *(const uint4*)&cB[r * 40 + kb];
      b[j].q[1] = *(const uint4*)&cB[r * 40 + kb + 16];
    }

#pragma unroll
    for (int i = 0; i < 2; ++i)
#pragma unroll
      for (int j = 0; j < 2; ++j)
        acc[i][j] = __builtin_amdgcn_wmma_f32_16x16x32_bf16(
            false, a[i].v, false, b[j].v, (short)0, acc[i][j], false, false);
    __syncthreads();                   // tiles consumed; buffer may be refilled
#if TDM_PATH
    buf ^= 1;
#endif
  }

  // ---- epilogue: C/D layout VGPR r -> M = r + 8*hi, N = lo ----
#pragma unroll
  for (int i = 0; i < 2; ++i) {
#pragma unroll
    for (int j = 0; j < 2; ++j) {
      const long col = colBase + wn * 32 + j * 16 + lo;
      const float bv = bias ? bias[col] : 0.0f;
#pragma unroll
      for (int r = 0; r < 8; ++r) {
        const long row  = rowBase + wm * 32 + i * 16 + hi * 8 + r;
        const float v   = acc[i][j][r] * alpha + bv;
        const long addr = dOff + row * ldD + col;
        if (Df32)  Df32[addr]  = v;
        if (Dbf16) Dbf16[addr] = (__bf16)v;
      }
    }
  }
}

// =====================================================================
// Row softmax (optionally causal): one 256-thread block per row.
// in f32 [batches][rowsPerBatch][n] -> out bf16 probabilities.
// =====================================================================
__global__ __launch_bounds__(256) void softmax_rows(
    const float* __restrict__ inp, __bf16* __restrict__ outp,
    int n, long rowsPerBatch, int causal)
{
  const long t = blockIdx.x;
  const long z = blockIdx.y;
  const float* row  = inp  + (z * rowsPerBatch + t) * n;
  __bf16*      orow = outp + (z * rowsPerBatch + t) * n;
  const int valid = causal ? (int)t + 1 : n;
  const int tid = threadIdx.x;
  __shared__ float red[256];

  float mx = -3.4e38f;
  for (int j = tid; j < valid; j += 256) mx = fmaxf(mx, row[j]);
  red[tid] = mx; __syncthreads();
  for (int s = 128; s > 0; s >>= 1) { if (tid < s) red[tid] = fmaxf(red[tid], red[tid + s]); __syncthreads(); }
  mx = red[0]; __syncthreads();

  float sum = 0.0f;
  for (int j = tid; j < valid; j += 256) sum += __expf(row[j] - mx);
  red[tid] = sum; __syncthreads();
  for (int s = 128; s > 0; s >>= 1) { if (tid < s) red[tid] += red[tid + s]; __syncthreads(); }
  const float inv = 1.0f / red[0];

  for (int j = tid; j < n; j += 256) {
    const float p = (j < valid) ? __expf(row[j] - mx) * inv : 0.0f;
    orow[j] = (__bf16)p;
  }
}

// ---------------- elementwise helpers ----------------
__global__ void k_f32_to_bf16(const float* __restrict__ in, __bf16* __restrict__ out, long n) {
  long i = (long)blockIdx.x * 256 + threadIdx.x;
  if (i < n) out[i] = (__bf16)in[i];
}

// in [R][Cc] f32 -> out [Cc][R] bf16  (weight pre-transpose)
__global__ void k_transpose_w(const float* __restrict__ in, __bf16* __restrict__ out, int R, int Cc) {
  long i = (long)blockIdx.x * 256 + threadIdx.x;
  long tot = (long)R * Cc;
  if (i >= tot) return;
  int r = (int)(i / Cc), c = (int)(i % Cc);
  out[(long)c * R + r] = (__bf16)in[i];
}

// qkv bf16 [Bn][L][3C] -> q/k/v bf16 [Bn][H][L][D]
__global__ void k_split_heads(const __bf16* __restrict__ qkv,
                              __bf16* __restrict__ q, __bf16* __restrict__ k, __bf16* __restrict__ v,
                              int Bn, int L, int Cc, int Hh, int Dd) {
  long i = (long)blockIdx.x * 256 + threadIdx.x;
  long tot = (long)Bn * L * 3 * Cc;
  if (i >= tot) return;
  int  c3 = (int)(i % (3 * Cc));
  long bl = i / (3 * Cc);
  int  l  = (int)(bl % L);
  int  bn = (int)(bl / L);
  int which = c3 / Cc, c = c3 % Cc, h = c / Dd, d = c % Dd;
  __bf16* dst = (which == 0) ? q : ((which == 1) ? k : v);
  dst[(((long)bn * Hh + h) * L + l) * Dd + d] = qkv[i];
}

// in [Z][L][D] bf16 -> out [Z][D][L] bf16  (V transpose for P@V GEMMs)
__global__ void k_transpose_LD(const __bf16* __restrict__ in, __bf16* __restrict__ out,
                               int Z, int L, int Dd) {
  long i = (long)blockIdx.x * 256 + threadIdx.x;
  long tot = (long)Z * L * Dd;
  if (i >= tot) return;
  int  d  = (int)(i % Dd);
  long zl = i / Dd;
  int  l  = (int)(zl % L);
  int  zz = (int)(zl / L);
  out[((long)zz * Dd + d) * L + l] = in[i];
}

// cval = cval1 + cval2 (in-place into c1); emit bf16 copies of cval and sval
__global__ void k_combine(const float* __restrict__ c2, float* __restrict__ c1,
                          const float* __restrict__ sval,
                          __bf16* __restrict__ c16, __bf16* __restrict__ s16, long n) {
  long i = (long)blockIdx.x * 256 + threadIdx.x;
  if (i >= n) return;
  float cv = c1[i] + c2[i];
  c1[i] = cv;
  c16[i] = (__bf16)cv;
  s16[i] = (__bf16)sval[i];
}

// mix = sigmoid(gs)*cval + sigmoid(gc)*sval  -> bf16
__global__ void k_gate(const float* __restrict__ gs, const float* __restrict__ gc,
                       const float* __restrict__ cval, const float* __restrict__ sval,
                       __bf16* __restrict__ mix, long n) {
  long i = (long)blockIdx.x * 256 + threadIdx.x;
  if (i >= n) return;
  float s = 1.0f / (1.0f + __expf(-gs[i]));
  float c = 1.0f / (1.0f + __expf(-gc[i]));
  mix[i] = (__bf16)(s * cval[i] + c * sval[i]);
}

// =====================================================================
// Host orchestration
// =====================================================================
extern "C" void kernel_launch(void* const* d_in, const int* in_sizes, int n_in,
                              void* d_out, int out_size, void* d_ws, size_t ws_size,
                              hipStream_t stream) {
  (void)in_sizes; (void)n_in; (void)out_size; (void)ws_size;
  const float* x      = (const float*)d_in[0];
  const float* y      = (const float*)d_in[1];
  /* d_in[2] = attn_x_mask (tril) -> implemented as causal predicate */
  const float* Wqkv_x = (const float*)d_in[3];
  const float* bqkv_x = (const float*)d_in[4];
  const float* Wqkv_y = (const float*)d_in[5];
  const float* bqkv_y = (const float*)d_in[6];
  const float* Wgs    = (const float*)d_in[7];
  const float* bgs    = (const float*)d_in[8];
  const float* Wgc    = (const float*)d_in[9];
  const float* bgc    = (const float*)d_in[10];
  const float* Wp     = (const float*)d_in[11];
  const float* bp     = (const float*)d_in[12];
  float* out = (float*)d_out;

  // ---- bump allocator over workspace ----
  char* w = (char*)d_ws;
  size_t off = 0;
  auto alloc = [&](size_t bytes) -> void* {
    void* p = w + off;
    off += (bytes + 255) & ~size_t(255);
    return p;
  };
  const long BTr = (long)kB * kT;           // 4096 rows
  const long BMr = (long)kB * kM;           // 1024 rows
  const long BH  = (long)kB * kH;           // 16 batches
  const long C3  = 3L * kC;                 // 1536

  __bf16* WqkvxT = (__bf16*)alloc(C3 * kC * 2);
  __bf16* WqkvyT = (__bf16*)alloc(C3 * kC * 2);
  __bf16* WgsT   = (__bf16*)alloc((long)kC * kC * 2);
  __bf16* WgcT   = (__bf16*)alloc((long)kC * kC * 2);
  __bf16* WpT    = (__bf16*)alloc((long)kC * kC * 2);
  __bf16* x16    = (__bf16*)alloc(BTr * kC * 2);
  __bf16* y16    = (__bf16*)alloc(BMr * kC * 2);
  __bf16* qkvx16 = (__bf16*)alloc(BTr * C3 * 2);
  __bf16* qkvy16 = (__bf16*)alloc(BMr * C3 * 2);
  __bf16* qx16   = (__bf16*)alloc(BH * kT * kD * 2);
  __bf16* kx16   = (__bf16*)alloc(BH * kT * kD * 2);
  __bf16* vx16   = (__bf16*)alloc(BH * kT * kD * 2);
  __bf16* qy16   = (__bf16*)alloc(BH * kM * kD * 2);
  __bf16* ky16   = (__bf16*)alloc(BH * kM * kD * 2);
  __bf16* vy16   = (__bf16*)alloc(BH * kM * kD * 2);
  __bf16* vxT    = (__bf16*)alloc(BH * kD * kT * 2);
  __bf16* vyT    = (__bf16*)alloc(BH * kD * kM * 2);
  float*  catt   = (float*) alloc(BH * kT * kM * 4);   // reused later for gate linears
  __bf16* catt16 = (__bf16*)alloc(BH * kT * kM * 2);
  __bf16* att2   = (__bf16*)alloc(BH * kT * kM * 2);   // (k_x . q_y) = catt_y2x^T
  __bf16* p16    = (__bf16*)alloc(BH * kT * kM * 2);
  float*  scoreT = (float*) alloc(BH * kT * kT * 4);   // prod, then self-attn scores
  __bf16* pp16   = (__bf16*)alloc(BH * kT * kT * 2);
  float*  cval1  = (float*) alloc(BTr * kC * 4);
  float*  cval2  = (float*) alloc(BTr * kC * 4);
  float*  sval   = (float*) alloc(BTr * kC * 4);
  __bf16* cval16 = (__bf16*)alloc(BTr * kC * 2);
  __bf16* sval16 = (__bf16*)alloc(BTr * kC * 2);
  __bf16* mix16  = (__bf16*)alloc(BTr * kC * 2);
  float*  gs_lin = catt;                                // dead after cross softmax
  float*  gc_lin = catt + BTr * kC;

  auto gblk = [](long n) { return dim3((unsigned)((n + 255) / 256)); };
  auto gemm = [&](const __bf16* A, const __bf16* BT, const float* bias, float alpha,
                  float* Df32, __bf16* Dbf16, long Mr, long N, int K,
                  long ldA, long ldBT, long ldD, long bsA, long bsBT,
                  long dOuter, long dInner, int dInnerCnt, int batches) {
    dim3 grid((unsigned)(Mr / 128), (unsigned)(N / 64), (unsigned)batches);
    wmma_gemm_bt<<<grid, 256, 0, stream>>>(A, BT, bias, alpha, Df32, Dbf16,
                                           K, ldA, ldBT, ldD, bsA, bsBT,
                                           dOuter, dInner, dInnerCnt);
  };

  // ---- 0. precision conversion / pre-transposes ----
  k_transpose_w<<<gblk((long)kC * C3), 256, 0, stream>>>(Wqkv_x, WqkvxT, kC, (int)C3);
  k_transpose_w<<<gblk((long)kC * C3), 256, 0, stream>>>(Wqkv_y, WqkvyT, kC, (int)C3);
  k_transpose_w<<<gblk((long)kC * kC), 256, 0, stream>>>(Wgs, WgsT, kC, kC);
  k_transpose_w<<<gblk((long)kC * kC), 256, 0, stream>>>(Wgc, WgcT, kC, kC);
  k_transpose_w<<<gblk((long)kC * kC), 256, 0, stream>>>(Wp,  WpT,  kC, kC);
  k_f32_to_bf16<<<gblk(BTr * kC), 256, 0, stream>>>(x, x16, BTr * kC);
  k_f32_to_bf16<<<gblk(BMr * kC), 256, 0, stream>>>(y, y16, BMr * kC);

  // ---- 1. QKV projections (bf16 out, f32 bias) ----
  gemm(x16, WqkvxT, bqkv_x, 1.0f, nullptr, qkvx16, BTr, C3, kC, kC, kC, C3, 0, 0, C3, 0, 1, 1);
  gemm(y16, WqkvyT, bqkv_y, 1.0f, nullptr, qkvy16, BMr, C3, kC, kC, kC, C3, 0, 0, C3, 0, 1, 1);
  k_split_heads<<<gblk(BTr * C3), 256, 0, stream>>>(qkvx16, qx16, kx16, vx16, kB, kT, kC, kH, kD);
  k_split_heads<<<gblk(BMr * C3), 256, 0, stream>>>(qkvy16, qy16, ky16, vy16, kB, kM, kC, kH, kD);
  k_transpose_LD<<<gblk(BH * kT * kD), 256, 0, stream>>>(vx16, vxT, (int)BH, kT, kD);
  k_transpose_LD<<<gblk(BH * kM * kD), 256, 0, stream>>>(vy16, vyT, (int)BH, kM, kD);

  // ---- 2. score GEMMs ----
  // catt_x2y[b,h,t,m] = SCALE * q_x . k_y   (f32 for softmax + bf16 for prod)
  gemm(qx16, ky16, nullptr, kSCALE, catt, catt16, kT, kM, kD,
       kD, kD, kM, (long)kT * kD, (long)kM * kD, (long)kT * kM, 0, 1, (int)BH);
  // att2[b,h,s,m] = SCALE * k_x . q_y  ( == catt_y2x^T, used as BT of prod GEMM )
  gemm(kx16, qy16, nullptr, kSCALE, nullptr, att2, kT, kM, kD,
       kD, kD, kM, (long)kT * kD, (long)kM * kD, (long)kT * kM, 0, 1, (int)BH);

  // ---- 3. cross attention x->y ----
  softmax_rows<<<dim3(kT, (unsigned)BH), 256, 0, stream>>>(catt, p16, kM, kT, 0);
  gemm(p16, vyT, nullptr, 1.0f, cval1, nullptr, kT, kD, kM,
       kM, kM, kC, (long)kT * kM, (long)kD * kM, (long)kT * kC, kD, kH, (int)BH);

  // ---- 4. product attention path ----
  // prod[b,h,t,s] = SCALE * sum_m catt16[t,m] * att2[s,m]
  gemm(catt16, att2, nullptr, kSCALE, scoreT, nullptr, kT, kT, kM,
       kM, kM, kT, (long)kT * kM, (long)kT * kM, (long)kT * kT, 0, 1, (int)BH);
  softmax_rows<<<dim3(kT, (unsigned)BH), 256, 0, stream>>>(scoreT, pp16, kT, kT, 1);
  gemm(pp16, vxT, nullptr, 1.0f, cval2, nullptr, kT, kD, kT,
       kT, kT, kC, (long)kT * kT, (long)kD * kT, (long)kT * kC, kD, kH, (int)BH);

  // ---- 5. self attention on x (reuse scoreT / pp16) ----
  gemm(qx16, kx16, nullptr, kSCALE, scoreT, nullptr, kT, kT, kD,
       kD, kD, kT, (long)kT * kD, (long)kT * kD, (long)kT * kT, 0, 1, (int)BH);
  softmax_rows<<<dim3(kT, (unsigned)BH), 256, 0, stream>>>(scoreT, pp16, kT, kT, 1);
  gemm(pp16, vxT, nullptr, 1.0f, sval, nullptr, kT, kD, kT,
       kT, kT, kC, (long)kT * kT, (long)kD * kT, (long)kT * kC, kD, kH, (int)BH);

  // ---- 6. gating + output projection ----
  k_combine<<<gblk(BTr * kC), 256, 0, stream>>>(cval2, cval1, sval, cval16, sval16, BTr * kC);
  gemm(sval16, WgsT, bgs, 1.0f, gs_lin, nullptr, BTr, kC, kC, kC, kC, kC, 0, 0, (long)kC, 0, 1, 1);
  gemm(cval16, WgcT, bgc, 1.0f, gc_lin, nullptr, BTr, kC, kC, kC, kC, kC, 0, 0, (long)kC, 0, 1, 1);
  k_gate<<<gblk(BTr * kC), 256, 0, stream>>>(gs_lin, gc_lin, cval1, sval, mix16, BTr * kC);
  gemm(mix16, WpT, bp, 1.0f, out, nullptr, BTr, kC, kC, kC, kC, kC, 0, 0, (long)kC, 0, 1, 1);
}
